// Deep_states_model_29996051595354
// MI455X (gfx1250) — compile-verified, matrix operand order checked by reading, not run on hardware
//
#include <hip/hip_runtime.h>
#include <hip/hip_bf16.h>
#include <cmath>

#define B_   16
#define T_   256
#define S_   16
#define H_   32
#define HL_  512
#define G4_  2048           // 4*HL
#define LOG2PI_F 1.8378770664093453f

typedef __attribute__((ext_vector_type(16))) __bf16 v16bf;
typedef __attribute__((ext_vector_type(8)))  float  v8f;
typedef __attribute__((ext_vector_type(2)))  float  v2f;

union V16U { v16bf v; uint4 q[2]; };

// Load a 16x32 bf16 WMMA operand fragment: two contiguous 16B chunks per lane.
__device__ __forceinline__ v16bf ld_ab(const __bf16* p0, const __bf16* p1) {
  V16U u;
  u.q[0] = *(const uint4*)p0;
  u.q[1] = *(const uint4*)p1;
  return u.v;
}

__device__ __forceinline__ v8f wmma_bf16(v16bf a, v16bf b, v8f c) {
  return __builtin_amdgcn_wmma_f32_16x16x32_bf16(false, a, false, b, (short)0, c, false, false);
}
__device__ __forceinline__ v8f wmma_f32(v2f a, v2f b, v8f c) {
  return __builtin_amdgcn_wmma_f32_16x16x4_f32(false, a, false, b, (short)0, c, false, false);
}

// wave-internal LDS producer->consumer sync (wave32 lockstep; just order + wait DS)
__device__ __forceinline__ void wave_sync() {
  __builtin_amdgcn_wave_barrier();
  asm volatile("s_wait_dscnt 0" ::: "memory");
  __builtin_amdgcn_wave_barrier();
}

// CDNA5 async global->LDS copy (ASYNCcnt-tracked), 16B per lane.
__device__ __forceinline__ void async_ld_b128(unsigned lds_off, const void* gaddr) {
  asm volatile("global_load_async_to_lds_b128 %0, %1, off" :: "v"(lds_off), "v"(gaddr) : "memory");
}
__device__ __forceinline__ void wait_async0() {
  asm volatile("s_wait_asynccnt 0" ::: "memory");
}

// ---------------------------------------------------------------------------
// Generic C[M,N](f32) = A(M,K)bf16 @ W(N,K)bf16^T + bias0[n] + bias1[n]
// One wave per 16x16 output tile (used for skinny GEMMs: N=16 or M=16).
// ---------------------------------------------------------------------------
__global__ void __launch_bounds__(32) gemm_bf16(
    const __bf16* __restrict__ A, long lda,
    const __bf16* __restrict__ W, long ldw,
    const float* __restrict__ bias0, const float* __restrict__ bias1,
    float* __restrict__ C, long ldc, int K)
{
  const int n0 = blockIdx.x * 16, m0 = blockIdx.y * 16;
  const int l = threadIdx.x, half = l >> 4, lm = l & 15;
  const __bf16* arow = A + (size_t)(m0 + lm) * lda;
  const __bf16* brow = W + (size_t)(n0 + lm) * ldw;
  v8f acc = {};
  for (int k0 = 0; k0 < K; k0 += 32) {
    const int kb = k0 + half * 8;
    v16bf a = ld_ab(arow + kb, arow + kb + 16);
    v16bf b = ld_ab(brow + kb, brow + kb + 16);
    acc = wmma_bf16(a, b, acc);
  }
#pragma unroll
  for (int r = 0; r < 8; r++) {
    const int row = m0 + r + half * 8, col = n0 + lm;
    float v = acc[r];
    if (bias0) v += bias0[col];
    if (bias1) v += bias1[col];
    C[(size_t)row * ldc + col] = v;
  }
}

// ---------------------------------------------------------------------------
// Tiled GEMM for the large panels: 64x64 block tile, BK=64, 8 waves,
// A/B staged to LDS via async global->LDS (double buffered), wmma from LDS.
// Requires M%64==0, N%64==0, K%64==0.
// ---------------------------------------------------------------------------
#define BM 64
#define BN 64
#define BKK 64

__global__ void __launch_bounds__(256) gemm_tiled(
    const __bf16* __restrict__ A, long lda,
    const __bf16* __restrict__ W, long ldw,
    const float* __restrict__ bias0, const float* __restrict__ bias1,
    float* __restrict__ C, long ldc, int K)
{
  __shared__ __bf16 sA[2][BM][BKK];
  __shared__ __bf16 sB[2][BN][BKK];
  const int m0 = blockIdx.y * BM, n0 = blockIdx.x * BN;
  const int tid = threadIdx.x;
  const int w = tid >> 5, lane = tid & 31, half = lane >> 4, lm = lane & 15;
  const int mi = w >> 1;        // 0..3  (16-row slab)
  const int nj0 = (w & 1) * 2;  // 0 or 2 (two 16-col tiles)

  auto issue = [&](int buf, int kbase) {
#pragma unroll
    for (int p = 0; p < 2; p++) {
      const int id = tid + p * 256;        // 0..511 chunk ids
      const int r = id >> 3, kc = (id & 7) * 8;  // row, element offset (16B chunks)
      async_ld_b128((unsigned)(uintptr_t)&sA[buf][r][kc],
                    (const void*)(A + (size_t)(m0 + r) * lda + kbase + kc));
      async_ld_b128((unsigned)(uintptr_t)&sB[buf][r][kc],
                    (const void*)(W + (size_t)(n0 + r) * ldw + kbase + kc));
    }
  };

  v8f acc[2];
  acc[0] = (v8f){};
  acc[1] = (v8f){};

  issue(0, 0);
  wait_async0();
  __syncthreads();

  const int nchunks = K / BKK;
  for (int c = 0; c < nchunks; c++) {
    const int buf = c & 1;
    if (c + 1 < nchunks) issue(buf ^ 1, (c + 1) * BKK);
#pragma unroll
    for (int kk = 0; kk < BKK; kk += 32) {
      const int kb = kk + half * 8;
      v16bf a = ld_ab(&sA[buf][mi * 16 + lm][kb], &sA[buf][mi * 16 + lm][kb + 16]);
#pragma unroll
      for (int j = 0; j < 2; j++) {
        v16bf b = ld_ab(&sB[buf][(nj0 + j) * 16 + lm][kb],
                        &sB[buf][(nj0 + j) * 16 + lm][kb + 16]);
        acc[j] = wmma_bf16(a, b, acc[j]);
      }
    }
    wait_async0();
    __syncthreads();
  }

#pragma unroll
  for (int j = 0; j < 2; j++) {
#pragma unroll
    for (int r = 0; r < 8; r++) {
      const int row = m0 + mi * 16 + r + half * 8;
      const int col = n0 + (nj0 + j) * 16 + lm;
      float v = acc[j][r];
      if (bias0) v += bias0[col];
      if (bias1) v += bias1[col];
      C[(size_t)row * ldc + col] = v;
    }
  }
}

// ---------------------------------------------------------------------------
// misc prep kernels
// ---------------------------------------------------------------------------
__global__ void cvt_bf16(__bf16* __restrict__ dst, const float* __restrict__ src, size_t n) {
  for (size_t i = (size_t)blockIdx.x * blockDim.x + threadIdx.x; i < n;
       i += (size_t)gridDim.x * blockDim.x)
    dst[i] = (__bf16)src[i];
}

// WFp[(s*1024 + r)*512 + k] = bf16(WF[(r*16 + s)*512 + k])
__global__ void perm_wf(__bf16* __restrict__ dst, const float* __restrict__ src) {
  const size_t n = (size_t)16384 * 512;
  for (size_t i = (size_t)blockIdx.x * blockDim.x + threadIdx.x; i < n;
       i += (size_t)gridDim.x * blockDim.x) {
    const int k = (int)(i & 511);
    const size_t row = i >> 9;
    const int s = (int)(row >> 10), r = (int)(row & 1023);
    dst[i] = (__bf16)src[(size_t)(r * 16 + s) * 512 + k];
  }
}

// bFp[s*1024 + r] = bF[r*16 + s]
__global__ void perm_bf(float* __restrict__ dst, const float* __restrict__ src) {
  for (int i = blockIdx.x * blockDim.x + threadIdx.x; i < 16384; i += gridDim.x * blockDim.x) {
    const int s = i >> 10, r = i & 1023;
    dst[i] = src[r * 16 + s];
  }
}

// Layer-0 input projection (K=4): xp[row][n] = feat[row,:4] . Wih0[n,:4] + bih0[n] + bhh0[n]
__global__ void xp0_kernel(const float* __restrict__ feat, const float* __restrict__ Wih0,
                           const float* __restrict__ bih, const float* __restrict__ bhh,
                           float* __restrict__ xp) {
  const size_t n_tot = (size_t)B_ * T_ * G4_;
  for (size_t i = (size_t)blockIdx.x * blockDim.x + threadIdx.x; i < n_tot;
       i += (size_t)gridDim.x * blockDim.x) {
    const int n = (int)(i & (G4_ - 1));
    const size_t row = i >> 11;
    const float* f = feat + row * 4;
    const float* w = Wih0 + (size_t)n * 4;
    xp[i] = f[0] * w[0] + f[1] * w[1] + f[2] * w[2] + f[3] * w[3] + bih[n] + bhh[n];
  }
}

// ---------------------------------------------------------------------------
// Persistent one-WGP LSTM layer. 32 waves; wave w owns hidden units [w*16,w*16+16).
// Per timestep each wave computes its 4 gate tiles (i,f,g,o) with WMMA
// (A = h_{t-1} in LDS bf16, B = Whh rows), then does the pointwise update with
// gate values held in the WMMA accumulators and c held in VGPRs.
// ---------------------------------------------------------------------------
__global__ void __launch_bounds__(1024) lstm_layer(
    const float* __restrict__ xp, const __bf16* __restrict__ Whh, __bf16* __restrict__ outbf)
{
  __shared__ __bf16 hbuf[2][B_][HL_];
  const int tid = threadIdx.x;
  const int w = tid >> 5, lane = tid & 31, half = lane >> 4, lm = lane & 15;
  const int hu = w * 16 + lm;

  for (int i = tid; i < 2 * B_ * HL_; i += 1024) ((__bf16*)hbuf)[i] = (__bf16)0.f;
  float c[8];
#pragma unroll
  for (int r = 0; r < 8; r++) c[r] = 0.f;
  __syncthreads();

  for (int t = 0; t < T_; t++) {
    const int pb = t & 1;
    v8f acc[4];
#pragma unroll
    for (int g = 0; g < 4; g++) acc[g] = (v8f){};

    for (int k0 = 0; k0 < HL_; k0 += 32) {
      const int kb = k0 + half * 8;
      v16bf a = ld_ab(&hbuf[pb][lm][kb], &hbuf[pb][lm][kb + 16]);
#pragma unroll
      for (int g = 0; g < 4; g++) {
        const __bf16* brow = Whh + ((size_t)(g * HL_ + w * 16 + lm)) * HL_ + kb;
        v16bf b = ld_ab(brow, brow + 16);
        acc[g] = wmma_bf16(a, b, acc[g]);
      }
    }
#pragma unroll
    for (int r = 0; r < 8; r++) {
      const int m = r + half * 8;  // batch index
      const size_t xbase = ((size_t)m * T_ + t) * G4_ + hu;
      const float gi = acc[0][r] + xp[xbase];
      const float gf = acc[1][r] + xp[xbase + HL_];
      const float gg = acc[2][r] + xp[xbase + 2 * HL_];
      const float go = acc[3][r] + xp[xbase + 3 * HL_];
      const float si = 1.f / (1.f + __expf(-gi));
      const float sf = 1.f / (1.f + __expf(-gf));
      const float so = 1.f / (1.f + __expf(-go));
      const float cc = sf * c[r] + si * tanhf(gg);
      c[r] = cc;
      const float hh = so * tanhf(cc);
      const __bf16 hb = (__bf16)hh;
      hbuf[pb ^ 1][m][hu] = hb;
      outbf[((size_t)m * T_ + t) * HL_ + hu] = hb;
    }
    __syncthreads();
  }
}

// ---------------------------------------------------------------------------
// 32x32 f32 matmul by one wave via V_WMMA_F32_16X16X4_F32.
// C = A @ B  (or A @ B^T when transB). All operands row-major 32x32 in LDS.
// ---------------------------------------------------------------------------
__device__ __forceinline__ void mat32_mul(const float* __restrict__ Am,
                                          const float* __restrict__ Bm,
                                          float* __restrict__ Co, bool transB, int lane)
{
  const int half = lane >> 4, lm = lane & 15;
#pragma unroll
  for (int ti = 0; ti < 2; ti++)
#pragma unroll
    for (int tj = 0; tj < 2; tj++) {
      v8f acc = {};
      const int mrow = ti * 16 + lm, ncol = tj * 16 + lm;
#pragma unroll
      for (int k0 = 0; k0 < 32; k0 += 4) {
        const int kk = k0 + half * 2;
        v2f av, bv;
        av.x = Am[mrow * 32 + kk];
        av.y = Am[mrow * 32 + kk + 1];
        if (transB) { bv.x = Bm[ncol * 32 + kk];  bv.y = Bm[ncol * 32 + kk + 1]; }
        else        { bv.x = Bm[kk * 32 + ncol];  bv.y = Bm[(kk + 1) * 32 + ncol]; }
        acc = wmma_f32(av, bv, acc);
      }
#pragma unroll
      for (int r = 0; r < 8; r++) Co[(ti * 16 + r + 8 * half) * 32 + ncol] = acc[r];
    }
}

// ---------------------------------------------------------------------------
// Kalman kernel: one workgroup per stock s. 8 waves / 256 threads.
// Per timestep: regenerate F (16 batches x 32x32) from out2 via bf16 WMMA
// (WF stays resident in the 192MB L2), then each wave updates 2 batches'
// (l,P) state with f32 WMMA + lane-parallel vector math.
// ---------------------------------------------------------------------------
__global__ void __launch_bounds__(256) kalman_kernel(
    const __bf16* __restrict__ out2, const __bf16* __restrict__ WFp,
    const float* __restrict__ bFp,
    const float* __restrict__ pa, const float* __restrict__ pg,
    const float* __restrict__ pb_, const float* __restrict__ psg,
    const float* __restrict__ mu0, const float* __restrict__ sig0,
    const float* __restrict__ price, float* __restrict__ outp)
{
  extern __shared__ float sm[];
  float* F   = sm;                 // 16*1024
  float* P   = F + 16 * 1024;      // 16*1024 (per-batch 32x32 covariance)
  float* lv  = P + 16 * 1024;      // 16*32
  float* scr = lv + 16 * 32;       // 8 waves * 2176

  const int s = blockIdx.x;
  const int tid = threadIdx.x, w = tid >> 5, lane = tid & 31, half = lane >> 4, lm = lane & 15;

  // init: l = mu0 slice, P = p0 * e0^T  (embeds the Hx1 initial p as HxH)
  for (int i = tid; i < 16 * 1024; i += 256) {
    const int b = i >> 10, ij = i & 1023, ii = ij >> 5, jj = ij & 31;
    P[i] = (jj == 0) ? sig0[(size_t)b * HL_ + ii * S_ + s] : 0.f;
  }
  for (int i = tid; i < 16 * 32; i += 256) {
    const int b = i >> 5, ii = i & 31;
    lv[i] = mu0[(size_t)b * HL_ + ii * S_ + s];
  }
  __syncthreads();

  float* T1   = scr + w * 2176;
  float* T2   = T1 + 1024;
  float* avec = T2 + 1024;
  float* vvec = avec + 32;
  float* wvec = vvec + 32;
  float* gv   = wvec + 32;

  for (int t = 0; t < T_; t++) {
    // ---- F = out2[:,t,:] @ WFp[s]^T + bFp[s] : 16 x 1024, wave w -> 8 tiles
    const __bf16* arow = out2 + ((size_t)lm * T_ + t) * HL_;
#pragma unroll 1
    for (int jt = 0; jt < 8; jt++) {
      const int n0 = (w * 8 + jt) * 16;
      const __bf16* brow = WFp + ((size_t)s * 1024 + n0 + lm) * HL_;
      v8f acc = {};
      for (int k0 = 0; k0 < HL_; k0 += 32) {
        const int kb = k0 + half * 8;
        v16bf a = ld_ab(arow + kb, arow + kb + 16);
        v16bf b = ld_ab(brow + kb, brow + kb + 16);
        acc = wmma_bf16(a, b, acc);
      }
#pragma unroll
      for (int r = 0; r < 8; r++) {
        const int m = r + 8 * half, col = n0 + lm;
        F[m * 1024 + col] = acc[r] + bFp[s * 1024 + col];
      }
    }
    __syncthreads();

    // ---- Kalman update, 2 batches per wave
    for (int bi = 0; bi < 2; bi++) {
      const int b = w * 2 + bi;
      float* Fb = F + b * 1024;
      float* Pb = P + b * 1024;
      float* lb = lv + b * 32;

      mat32_mul(Fb, Pb, T1, false, lane);   // T1 = F @ P
      wave_sync();
      mat32_mul(T1, Pb, T2, true, lane);    // T2 = (F @ P) @ P^T

      // l_new = F @ l (uses old l)
      float nl = 0.f;
      for (int k = 0; k < 32; k++) nl += Fb[lane * 32 + k] * lb[k];

      const size_t rbase = (size_t)b * T_ + t;
      gv[lane]   = pg[rbase * HL_ + lane * S_ + s];
      avec[lane] = pa[rbase * HL_ + lane * S_ + s];
      wave_sync();

      // P = T2 + g g^T
      const float gi_ = gv[lane];
      for (int j = 0; j < 32; j++) Pb[lane * 32 + j] = T2[lane * 32 + j] + gi_ * gv[j];
      lb[lane] = nl;
      wave_sync();

      // z_pred = a . l + b ;  v = P a^T
      float zp = pb_[rbase * S_ + s];
      for (int k = 0; k < 32; k++) zp += avec[k] * lb[k];
      float vv = 0.f;
      for (int j = 0; j < 32; j++) vv += Pb[lane * 32 + j] * avec[j];
      vvec[lane] = vv;
      wave_sync();

      // s = |a . v + sg^2| with NaN guards
      const float sg = psg[rbase * S_ + s];
      float sval = sg * sg;
      for (int k = 0; k < 32; k++) sval += avec[k] * vvec[k];
      sval = fabsf(sval);
      if (sval != sval) sval = 0.001f;
      if (zp != zp) zp = 0.001f;

      const float kv = vvec[lane] / sval;         // k = P a^T / s
      const float z = price[rbase * S_ + s];
      const float resid = z - zp;

      // w = a @ P (pre-update P)
      float wj = 0.f;
      for (int i2 = 0; i2 < 32; i2++) wj += avec[i2] * Pb[i2 * 32 + lane];
      wvec[lane] = wj;
      wave_sync();

      lb[lane] += kv * resid;                     // l += k (z - z_pred)
      for (int j = 0; j < 32; j++) Pb[lane * 32 + j] -= kv * wvec[j];  // P -= k w

      if (t == T_ - 1 && lane == 0) {
        const float q = resid / sval;
        outp[b * S_ + s] = -0.5f * q * q - __logf(sval) - 0.5f * LOG2PI_F;
      }
      wave_sync();
    }
    __syncthreads();
  }
}

// ---------------------------------------------------------------------------
// host orchestration
// ---------------------------------------------------------------------------
extern "C" void kernel_launch(void* const* d_in, const int* in_sizes, int n_in,
                              void* d_out, int out_size, void* d_ws, size_t ws_size,
                              hipStream_t stream)
{
  (void)in_sizes; (void)n_in; (void)out_size; (void)ws_size;
  const float* feat  = (const float*)d_in[0];
  const float* price = (const float*)d_in[1];
  const float* Wih[3] = {(const float*)d_in[2], (const float*)d_in[6],  (const float*)d_in[10]};
  const float* Whh[3] = {(const float*)d_in[3], (const float*)d_in[7],  (const float*)d_in[11]};
  const float* bih[3] = {(const float*)d_in[4], (const float*)d_in[8],  (const float*)d_in[12]};
  const float* bhh[3] = {(const float*)d_in[5], (const float*)d_in[9],  (const float*)d_in[13]};
  const float* WF   = (const float*)d_in[14]; const float* bF   = (const float*)d_in[15];
  const float* Wa   = (const float*)d_in[16]; const float* ba   = (const float*)d_in[17];
  const float* Wb   = (const float*)d_in[18]; const float* bb   = (const float*)d_in[19];
  const float* Wg   = (const float*)d_in[20]; const float* bg   = (const float*)d_in[21];
  const float* Wsig = (const float*)d_in[22]; const float* bsig = (const float*)d_in[23];
  const float* Wmu0 = (const float*)d_in[24]; const float* bmu0 = (const float*)d_in[25];
  const float* Wsg0 = (const float*)d_in[26]; const float* bsg0 = (const float*)d_in[27];

  char* wsp = (char*)d_ws;
  auto alloc = [&](size_t bytes) -> char* {
    char* p = wsp; wsp += (bytes + 255) & ~(size_t)255; return p;
  };
  const size_t MT = (size_t)B_ * T_;
  __bf16* OUT0 = (__bf16*)alloc(MT * HL_ * 2);
  __bf16* OUT1 = (__bf16*)alloc(MT * HL_ * 2);
  __bf16* OUT2 = (__bf16*)alloc(MT * HL_ * 2);
  float*  XP   = (float*) alloc(MT * G4_ * 4);
  __bf16* WIH1 = (__bf16*)alloc((size_t)G4_ * HL_ * 2);
  __bf16* WIH2 = (__bf16*)alloc((size_t)G4_ * HL_ * 2);
  __bf16* WHHb[3];
  for (int i = 0; i < 3; i++) WHHb[i] = (__bf16*)alloc((size_t)G4_ * HL_ * 2);
  __bf16* WFP  = (__bf16*)alloc((size_t)16384 * HL_ * 2);
  float*  BFP  = (float*) alloc(16384 * 4);
  __bf16* WAb  = (__bf16*)alloc((size_t)HL_ * HL_ * 2);
  __bf16* WGb  = (__bf16*)alloc((size_t)HL_ * HL_ * 2);
  __bf16* WBb  = (__bf16*)alloc((size_t)S_ * HL_ * 2);
  __bf16* WSGb = (__bf16*)alloc((size_t)S_ * HL_ * 2);
  __bf16* WMU0b= (__bf16*)alloc((size_t)HL_ * HL_ * 2);
  __bf16* WSG0b= (__bf16*)alloc((size_t)HL_ * HL_ * 2);
  float*  PA   = (float*) alloc(MT * HL_ * 4);
  float*  PG   = (float*) alloc(MT * HL_ * 4);
  float*  PB   = (float*) alloc(MT * S_ * 4);
  float*  PSG  = (float*) alloc(MT * S_ * 4);
  float*  MU0  = (float*) alloc((size_t)B_ * HL_ * 4);
  float*  SIG0 = (float*) alloc((size_t)B_ * HL_ * 4);

  auto cvt = [&](__bf16* dst, const float* src, size_t n) {
    cvt_bf16<<<(unsigned)((n + 255) / 256), 256, 0, stream>>>(dst, src, n);
  };
  cvt(WIH1, Wih[1], (size_t)G4_ * HL_);
  cvt(WIH2, Wih[2], (size_t)G4_ * HL_);
  for (int i = 0; i < 3; i++) cvt(WHHb[i], Whh[i], (size_t)G4_ * HL_);
  cvt(WAb, Wa, (size_t)HL_ * HL_);
  cvt(WGb, Wg, (size_t)HL_ * HL_);
  cvt(WBb, Wb, (size_t)S_ * HL_);
  cvt(WSGb, Wsig, (size_t)S_ * HL_);
  cvt(WMU0b, Wmu0, (size_t)HL_ * HL_);
  cvt(WSG0b, Wsg0, (size_t)HL_ * HL_);
  perm_wf<<<8192, 256, 0, stream>>>(WFP, WF);
  perm_bf<<<64, 256, 0, stream>>>(BFP, bF);

  // layer 0
  xp0_kernel<<<(unsigned)((MT * G4_ + 255) / 256), 256, 0, stream>>>(feat, Wih[0], bih[0], bhh[0], XP);
  lstm_layer<<<1, 1024, 0, stream>>>(XP, WHHb[0], OUT0);

  // layer 1 (async-LDS tiled GEMM for the 4096x2048x512 panel)
  gemm_tiled<<<dim3(G4_ / BN, (unsigned)(MT / BM)), 256, 0, stream>>>(
      OUT0, HL_, WIH1, HL_, bih[1], bhh[1], XP, G4_, HL_);
  lstm_layer<<<1, 1024, 0, stream>>>(XP, WHHb[1], OUT1);

  // layer 2
  gemm_tiled<<<dim3(G4_ / BN, (unsigned)(MT / BM)), 256, 0, stream>>>(
      OUT1, HL_, WIH2, HL_, bih[2], bhh[2], XP, G4_, HL_);
  lstm_layer<<<1, 1024, 0, stream>>>(XP, WHHb[2], OUT2);

  // projections (a, g: N=512 tiled; b, sig: N=16 skinny)
  gemm_tiled<<<dim3(HL_ / BN, (unsigned)(MT / BM)), 256, 0, stream>>>(
      OUT2, HL_, WAb, HL_, ba, nullptr, PA, HL_, HL_);
  gemm_tiled<<<dim3(HL_ / BN, (unsigned)(MT / BM)), 256, 0, stream>>>(
      OUT2, HL_, WGb, HL_, bg, nullptr, PG, HL_, HL_);
  gemm_bf16<<<dim3(S_ / 16, (unsigned)(MT / 16)), 32, 0, stream>>>(
      OUT2, HL_, WBb, HL_, bb, nullptr, PB, S_, HL_);
  gemm_bf16<<<dim3(S_ / 16, (unsigned)(MT / 16)), 32, 0, stream>>>(
      OUT2, HL_, WSGb, HL_, bsig, nullptr, PSG, S_, HL_);

  // initial state from h0 = out2[:, 0, :]  (row stride T*HL picks t=0 rows)
  gemm_bf16<<<dim3(HL_ / 16, B_ / 16), 32, 0, stream>>>(
      OUT2, (long)T_ * HL_, WMU0b, HL_, bmu0, nullptr, MU0, HL_, HL_);
  gemm_bf16<<<dim3(HL_ / 16, B_ / 16), 32, 0, stream>>>(
      OUT2, (long)T_ * HL_, WSG0b, HL_, bsg0, nullptr, SIG0, HL_, HL_);

  // Kalman filter: one workgroup per stock
  const size_t smem = (size_t)(16 * 1024 + 16 * 1024 + 16 * 32 + 8 * 2176) * 4;
  kalman_kernel<<<S_, 256, smem, stream>>>(
      OUT2, WFP, BFP, PA, PG, PB, PSG, MU0, SIG0, price, (float*)d_out);
}
